// SpectralConv2d_13391708029589
// MI455X (gfx1250) — matrix operand order repaired and checked by reading, not compiled, not run on hardware
//
#include <hip/hip_runtime.h>
#include <math.h>

typedef __attribute__((ext_vector_type(16))) _Float16 v16h;
typedef __attribute__((ext_vector_type(8)))  _Float16 v8h;
typedef __attribute__((ext_vector_type(8)))  float    v8f;
typedef __attribute__((ext_vector_type(4)))  float    v4f;

#define NB   16
#define MS   256
#define NS   256
#define CI   64
#define CO   64
#define KMOD 32
#define FFH  256
#define LN_EPS 1e-5f

#define WMMA(a, b, c) __builtin_amdgcn_wmma_f32_16x16x32_f16(false, (a), false, (b), (short)0, (c), false, false)

// ---- WMMA fragment loaders ---------------------------------------------
// A (16x32 f16): lane holds row m=lane&15; halves h -> k = (lane<16?0:16)+h.
// Row-major [rows][ld].
static __device__ __forceinline__ v16h ld_afrag(const _Float16* p, int ld, int row0, int k0) {
  int lane = threadIdx.x & 31;
  int row  = row0 + (lane & 15);
  int hi8  = (lane & 16) ? 16 : 0;
  const _Float16* base = p + row * ld + k0 + hi8;
  return *(const v16h*)(base);
}

// B (32x16 f16): lane holds col n=lane&15; halves h -> k = (lane<16?0:16)+h.
// Stored column-major (Bt[n][ld=K]) -> one contiguous 32B load.
static __device__ __forceinline__ v16h ld_bfrag(const _Float16* p, int ld, int col0, int k0) {
  int lane = threadIdx.x & 31;
  int col  = col0 + (lane & 15);
  int ko   = k0 + ((lane & 16) ? 16 : 0);
  return *(const v16h*)(p + col * ld + ko);
}

// ---- Precompute kernels ------------------------------------------------
// Ft[col][n], col<32: cos(2*pi*k*n/256)/16 ; col>=32 (k=col-32): -sin(...)/16
// Gt[n][kcat], kcat<32: c_k*cos/16 ; kcat>=32: -c_k*sin/16, c_0=1 else 2
__global__ void k_prep_dft(_Float16* __restrict__ Ft, _Float16* __restrict__ Gt) {
  int idx = blockIdx.x * blockDim.x + threadIdx.x;
  if (idx >= 64 * 256) return;
  const float inv = 1.0f / 16.0f;                  // 1/sqrt(256), ortho norm
  const float w   = 6.283185307179586f / 256.0f;
  int col = idx >> 8, n = idx & 255;
  int k = col & 31;
  float ang = w * (float)(k * n);
  Ft[idx] = (_Float16)((col < 32) ? cosf(ang) * inv : -sinf(ang) * inv);
  int n2 = idx >> 6, kc = idx & 63;
  int k2 = kc & 31;
  float ck = (k2 == 0) ? 1.0f : 2.0f;
  float a2 = w * (float)(k2 * n2);
  Gt[idx] = (_Float16)((kc < 32) ? ck * cosf(a2) * inv : -ck * sinf(a2) * inv);
}

// Complex mode-mix expanded to real 128x128 per mode, stored Bt: Wpt[k][ocat][c]
__global__ void k_prep_w(const float* __restrict__ wre, const float* __restrict__ wim,
                         _Float16* __restrict__ Wpt) {
  int idx = blockIdx.x * blockDim.x + threadIdx.x;
  if (idx >= KMOD * 128 * 128) return;
  int c    = idx & 127;
  int ocat = (idx >> 7) & 127;
  int k    = idx >> 14;
  int i = c & 63, o = ocat & 63;
  int widx = (i * CO + o) * KMOD + k;          // w[C][OUT][KMOD]
  float re = wre[widx], im = wim[widx];
  float v;
  if (c < 64) v = (ocat < 64) ? re : im;       // rows: re-part of modes
  else        v = (ocat < 64) ? -im : re;      // rows: im-part of modes
  Wpt[idx] = (_Float16)v;
}

// FF weights transposed to Bt layouts: W1t[hid][in], W2t[out][hid]
__global__ void k_prep_ff(const float* __restrict__ w1, const float* __restrict__ w2,
                          _Float16* __restrict__ w1t, _Float16* __restrict__ w2t) {
  int idx = blockIdx.x * blockDim.x + threadIdx.x;
  if (idx >= 16384) return;
  int i = idx & 63, h = idx >> 6;
  w1t[idx] = (_Float16)w1[i * FFH + h];
  int h2 = idx & 255, o = idx >> 8;
  w2t[idx] = (_Float16)w2[h2 * CO + o];
}

// ---- Stage A: truncated forward DFT as GEMM ----------------------------
// Per (b, r): Mo[i][kcat64] = X(64 x 256) @ F(256 x 64).  axis=0: r=m (DFT
// along n); axis=1: r=n (DFT along m).  Writes Mbuf[b][k][r][c128], c = i|64+i.
__global__ void __launch_bounds__(256) k_dft_fwd(const float* __restrict__ x,
    const _Float16* __restrict__ Ft, _Float16* __restrict__ Mbuf, int axis) {
  __shared__ __align__(32) _Float16 A_lds[64 * 32];   // [i][t]
  int b = blockIdx.x >> 8;
  int r = blockIdx.x & 255;
  int tid = threadIdx.x, wave = tid >> 5, lane = tid & 31;
  v8f acc[2] = {};
  for (int kc = 0; kc < 8; ++kc) {
#pragma unroll
    for (int j = 0; j < 2; ++j) {               // stage x -> f16 transpose
      int e = tid * 8 + j * 4;
      int i = e & 63, t = e >> 6;
      int ta = kc * 32 + t;
      int mm = axis ? ta : r;
      int nn = axis ? r : ta;
      v4f v = *(const v4f*)(x + (((size_t)(b * MS + mm) * NS + nn) << 6) + i);
#pragma unroll
      for (int q = 0; q < 4; ++q) A_lds[(i + q) * 32 + t] = (_Float16)v[q];
    }
    __syncthreads();
#pragma unroll
    for (int tt = 0; tt < 2; ++tt) {            // 16 tiles / 8 waves
      int id = wave * 2 + tt;
      int ti = id >> 2, tj = id & 3;
      v16h a  = ld_afrag(A_lds, 32, ti * 16, 0);
      v16h bb = ld_bfrag(Ft, 256, tj * 16, kc * 32);
      acc[tt] = WMMA(a, bb, acc[tt]);
    }
    __syncthreads();
  }
#pragma unroll
  for (int tt = 0; tt < 2; ++tt) {
    int id = wave * 2 + tt;
    int ti = id >> 2, tj = id & 3;
    int kcat = tj * 16 + (lane & 15);
    int k    = kcat & 31;
    int c0   = ((kcat < 32) ? 0 : 64) + ti * 16 + ((lane & 16) ? 8 : 0);
    v8h hv;
#pragma unroll
    for (int rr = 0; rr < 8; ++rr) hv[rr] = (_Float16)acc[tt][rr];
    *(v8h*)(Mbuf + (((size_t)(b * KMOD + k) * 256 + r) << 7) + c0) = hv;
  }
}

// ---- Stage B: per-mode complex channel mixing as real GEMM -------------
// Per (b,k,mq): [64 rows x 128 c] @ Wpt[k] (128x128) -> Ybuf[b][row][o][kcat]
__global__ void __launch_bounds__(256) k_mix(const _Float16* __restrict__ Mbuf,
    const _Float16* __restrict__ Wpt, _Float16* __restrict__ Ybuf) {
  int mq = blockIdx.x & 3;
  int k  = (blockIdx.x >> 2) & 31;
  int b  = blockIdx.x >> 7;
  int tid = threadIdx.x, wave = tid >> 5, lane = tid & 31;
  const _Float16* Abase = Mbuf + ((size_t)(b * KMOD + k) * 256 + mq * 64) * 128;
  const _Float16* Bbase = Wpt + (size_t)k * 128 * 128;
  v8f acc[4] = {};
  for (int kc = 0; kc < 4; ++kc) {
#pragma unroll
    for (int tt = 0; tt < 4; ++tt) {            // 32 tiles / 8 waves
      int id = wave * 4 + tt;
      int ti = id >> 3, tj = id & 7;
      v16h a  = ld_afrag(Abase, 128, ti * 16, kc * 32);
      v16h bb = ld_bfrag(Bbase, 128, tj * 16, kc * 32);
      acc[tt] = WMMA(a, bb, acc[tt]);
    }
  }
#pragma unroll
  for (int tt = 0; tt < 4; ++tt) {
    int id = wave * 4 + tt;
    int ti = id >> 3, tj = id & 7;
    int ocat = tj * 16 + (lane & 15);
    int o    = ocat & 63;
    int kcat = k + ((ocat & 64) ? 32 : 0);
    int m0   = mq * 64 + ti * 16 + ((lane & 16) ? 8 : 0);
#pragma unroll
    for (int rr = 0; rr < 8; ++rr)
      Ybuf[(((size_t)(b * 256 + m0 + rr) * 64 + o) << 6) + kcat] = (_Float16)acc[tt][rr];
  }
}

// ---- Stage C: truncated inverse DFT as GEMM, accumulate both axes ------
// Per (b,r): D(64 o x 256 cols) = Ybuf[b][r] (64x64) @ Gt-as-B (64x256)
__global__ void __launch_bounds__(256) k_idft(const _Float16* __restrict__ Ybuf,
    const _Float16* __restrict__ Gt, _Float16* __restrict__ Acc, int axis, int add) {
  int b = blockIdx.x >> 8;
  int r = blockIdx.x & 255;
  int tid = threadIdx.x, wave = tid >> 5, lane = tid & 31;
  const _Float16* Abase = Ybuf + (size_t)(b * 256 + r) * 64 * 64;
  v8f acc[8] = {};
#pragma unroll
  for (int kc = 0; kc < 2; ++kc) {
#pragma unroll
    for (int tt = 0; tt < 8; ++tt) {            // 64 tiles / 8 waves
      int id = wave * 8 + tt;
      int ti = id >> 4, tj = id & 15;
      v16h a  = ld_afrag(Abase, 64, ti * 16, kc * 32);
      v16h bb = ld_bfrag(Gt, 64, tj * 16, kc * 32);
      acc[tt] = WMMA(a, bb, acc[tt]);
    }
  }
#pragma unroll
  for (int tt = 0; tt < 8; ++tt) {
    int id = wave * 8 + tt;
    int ti = id >> 4, tj = id & 15;
    int col = tj * 16 + (lane & 15);
    int o0  = ti * 16 + ((lane & 16) ? 8 : 0);
    int mm = axis ? col : r;
    int nn = axis ? r : col;
    _Float16* p = Acc + ((((size_t)(b * MS + mm) * NS + nn) << 6) + o0);
    v8h nv;
    if (add) {
      v8h old = *(v8h*)p;
#pragma unroll
      for (int rr = 0; rr < 8; ++rr) nv[rr] = (_Float16)((float)old[rr] + acc[tt][rr]);
    } else {
#pragma unroll
      for (int rr = 0; rr < 8; ++rr) nv[rr] = (_Float16)acc[tt][rr];
    }
    *(v8h*)p = nv;
  }
}

// ---- FF (64->256 ReLU ->64) + LayerNorm, fused, 64 tokens per block ----
__global__ void __launch_bounds__(256) k_ff(const _Float16* __restrict__ Acc,
    const _Float16* __restrict__ W1t, const float* __restrict__ b1,
    const _Float16* __restrict__ W2t, const float* __restrict__ b2,
    const float* __restrict__ lng, const float* __restrict__ lnb,
    float* __restrict__ out) {
  __shared__ __align__(32) _Float16 H[64 * 256];   // hidden, post-ReLU
  __shared__ float Y[64 * 64];                     // pre-LN output
  int tok0 = blockIdx.x * 64;
  int tid = threadIdx.x, wave = tid >> 5, lane = tid & 31;
  const _Float16* Abase = Acc + (size_t)tok0 * 64;
  {                                               // GEMM1: [64x64]@[64x256]
    v8f acc[8] = {};
#pragma unroll
    for (int kc = 0; kc < 2; ++kc) {
#pragma unroll
      for (int tt = 0; tt < 8; ++tt) {
        int id = wave * 8 + tt;
        int ti = id >> 4, tj = id & 15;
        v16h a  = ld_afrag(Abase, 64, ti * 16, kc * 32);
        v16h bb = ld_bfrag(W1t, 64, tj * 16, kc * 32);
        acc[tt] = WMMA(a, bb, acc[tt]);
      }
    }
#pragma unroll
    for (int tt = 0; tt < 8; ++tt) {
      int id = wave * 8 + tt;
      int ti = id >> 4, tj = id & 15;
      int hcol = tj * 16 + (lane & 15);
      float bias = b1[hcol];
      int t0 = ti * 16 + ((lane & 16) ? 8 : 0);
#pragma unroll
      for (int rr = 0; rr < 8; ++rr) {
        float v = acc[tt][rr] + bias;
        H[(t0 + rr) * 256 + hcol] = (_Float16)(v > 0.0f ? v : 0.0f);
      }
    }
  }
  __syncthreads();
  {                                               // GEMM2: [64x256]@[256x64]
    v8f acc[2] = {};
    for (int kc = 0; kc < 8; ++kc) {
#pragma unroll
      for (int tt = 0; tt < 2; ++tt) {
        int id = wave * 2 + tt;
        int ti = id >> 2, tj = id & 3;
        v16h a  = ld_afrag(H, 256, ti * 16, kc * 32);
        v16h bb = ld_bfrag(W2t, 256, tj * 16, kc * 32);
        acc[tt] = WMMA(a, bb, acc[tt]);
      }
    }
#pragma unroll
    for (int tt = 0; tt < 2; ++tt) {
      int id = wave * 2 + tt;
      int ti = id >> 2, tj = id & 3;
      int oc = tj * 16 + (lane & 15);
      float bias = b2[oc];
      int t0 = ti * 16 + ((lane & 16) ? 8 : 0);
#pragma unroll
      for (int rr = 0; rr < 8; ++rr) Y[(t0 + rr) * 64 + oc] = acc[tt][rr] + bias;
    }
  }
  __syncthreads();
  if (tid < 64) {                                 // LayerNorm (f32)
    const float* yr = &Y[tid * 64];
    float mu = 0.0f;
    for (int o = 0; o < 64; ++o) mu += yr[o];
    mu *= (1.0f / 64.0f);
    float var = 0.0f;
    for (int o = 0; o < 64; ++o) { float d = yr[o] - mu; var += d * d; }
    var *= (1.0f / 64.0f);
    float inv = rsqrtf(var + LN_EPS);
    float* op = out + (size_t)(tok0 + tid) * 64;
    for (int o = 0; o < 64; ++o) op[o] = (yr[o] - mu) * inv * lng[o] + lnb[o];
  }
}

// ---- Launch ------------------------------------------------------------
extern "C" void kernel_launch(void* const* d_in, const int* in_sizes, int n_in,
                              void* d_out, int out_size, void* d_ws, size_t ws_size,
                              hipStream_t stream) {
  (void)in_sizes; (void)n_in; (void)out_size; (void)ws_size;
  const float* x     = (const float*)d_in[0];
  const float* w1re  = (const float*)d_in[1];
  const float* w1im  = (const float*)d_in[2];
  const float* w2re  = (const float*)d_in[3];
  const float* w2im  = (const float*)d_in[4];
  const float* ffw1  = (const float*)d_in[5];
  const float* ffb1  = (const float*)d_in[6];
  const float* ffw2  = (const float*)d_in[7];
  const float* ffb2  = (const float*)d_in[8];
  const float* lng   = (const float*)d_in[9];
  const float* lnb   = (const float*)d_in[10];
  float* out = (float*)d_out;

  char* ws = (char*)d_ws;
  size_t oFt  = 0;                       // 64*256 f16      = 32 KB
  size_t oGt  = oFt  + 32768;            // 256*64 f16      = 32 KB
  size_t oW1p = oGt  + 32768;            // 32*128*128 f16  = 1 MB
  size_t oW2p = oW1p + 1048576;          // 1 MB
  size_t oF1t = oW2p + 1048576;          // 256*64 f16
  size_t oF2t = oF1t + 32768;            // 64*256 f16
  size_t oMb  = oF2t + 32768;            // 16*32*256*128 f16 = 32 MB
  size_t oYb  = oMb  + 33554432;         // 16*256*64*64 f16  = 32 MB
  size_t oAcc = oYb  + 33554432;         // 16*256*256*64 f16 = 128 MB
  _Float16* Ft  = (_Float16*)(ws + oFt);
  _Float16* Gt  = (_Float16*)(ws + oGt);
  _Float16* W1p = (_Float16*)(ws + oW1p);
  _Float16* W2p = (_Float16*)(ws + oW2p);
  _Float16* F1t = (_Float16*)(ws + oF1t);
  _Float16* F2t = (_Float16*)(ws + oF2t);
  _Float16* Mb  = (_Float16*)(ws + oMb);
  _Float16* Yb  = (_Float16*)(ws + oYb);
  _Float16* Acc = (_Float16*)(ws + oAcc);

  k_prep_dft<<<64, 256, 0, stream>>>(Ft, Gt);
  k_prep_w<<<2048, 256, 0, stream>>>(w1re, w1im, W1p);
  k_prep_w<<<2048, 256, 0, stream>>>(w2re, w2im, W2p);
  k_prep_ff<<<64, 256, 0, stream>>>(ffw1, ffw2, F1t, F2t);

  // y-branch (DFT along n)
  k_dft_fwd<<<NB * MS, 256, 0, stream>>>(x, Ft, Mb, 0);
  k_mix<<<NB * KMOD * 4, 256, 0, stream>>>(Mb, W1p, Yb);
  k_idft<<<NB * MS, 256, 0, stream>>>(Yb, Gt, Acc, 0, 0);

  // x-branch (DFT along m), accumulates into Acc
  k_dft_fwd<<<NB * NS, 256, 0, stream>>>(x, Ft, Mb, 1);
  k_mix<<<NB * KMOD * 4, 256, 0, stream>>>(Mb, W2p, Yb);
  k_idft<<<NB * NS, 256, 0, stream>>>(Yb, Gt, Acc, 1, 1);

  // fused FF + LayerNorm
  k_ff<<<(NB * MS * NS) / 64, 256, 0, stream>>>(Acc, F1t, ffb1, F2t, ffb2, lng, lnb, out);
}
